// Attention_39994735460566
// MI455X (gfx1250) — matrix operand order repaired and checked by reading, not compiled, not run on hardware
//
#include <hip/hip_runtime.h>
#include <hip/hip_bf16.h>

typedef _Float16 f16;
typedef __attribute__((ext_vector_type(16))) _Float16 v16h;
typedef __attribute__((ext_vector_type(8)))  _Float16 v8h;
typedef __attribute__((ext_vector_type(8)))  float    v8f;

#define DIM    768
#define HEADS  12
#define DHEAD  64
#define SEQ    2048
#define BATCH  4
#define ROWS   (BATCH * SEQ)   // 8192
#define QKV_N  (3 * DIM)       // 2304

// ---------------------------------------------------------------------------
// WMMA fragment helpers (CDNA5 16x16x32 f16 layouts, wave32)
// A (16x32, MxK): lane = {row = lane&15, kh = lane>>4}; holds K = kh*8..kh*8+7
//                 and K = 16+kh*8..16+kh*8+7  (two 16B chunks)
// B (32x16, KxN): lane = {col = lane&15, kh = lane>>4}; holds K = kh*8..kh*8+7
//                 and K = 16+kh*8..16+kh*8+7  (two 16B chunks)
// C/D (16x16 f32): VGPR r, lane L -> element (row = r + 8*(L>>4), col = L&15)
// ---------------------------------------------------------------------------
static __device__ inline v16h ld_frag_a(const f16* base, int ld) {
  const int lane = threadIdx.x & 31;
  const int row  = lane & 15;
  const int kh   = lane >> 4;
  const f16* p = base + row * ld + kh * 8;
  v8h lo = *(const v8h*)(p);
  v8h hi = *(const v8h*)(p + 16);
  v16h a;
#pragma unroll
  for (int i = 0; i < 8; ++i) { a[i] = lo[i]; a[i + 8] = hi[i]; }
  return a;
}

static __device__ inline v16h ld_frag_b(const f16* base, int ld) {
  const int lane = threadIdx.x & 31;
  const int col  = lane & 15;
  const int kh   = lane >> 4;
  const f16* p = base + col * ld + kh * 8;
  v8h lo = *(const v8h*)(p);
  v8h hi = *(const v8h*)(p + 16);
  v16h b;
#pragma unroll
  for (int i = 0; i < 8; ++i) { b[i] = lo[i]; b[i + 8] = hi[i]; }
  return b;
}

static __device__ inline v8f wmma32(v16h a, v16h b, v8f c) {
  return __builtin_amdgcn_wmma_f32_16x16x32_f16(
      /*neg_a=*/false, a, /*neg_b=*/false, b,
      /*c_mod=*/(short)0, c, /*reuse_a=*/false, /*reuse_b=*/false);
}

static __device__ inline v8f vzero8f() {
  v8f z;
#pragma unroll
  for (int i = 0; i < 8; ++i) z[i] = 0.0f;
  return z;
}

// ---------------------------------------------------------------------------
// Kernel 1: weight convert + transpose  in[K][N] f32 -> out[N][K] f16
// ---------------------------------------------------------------------------
__global__ void transpose_f16_kernel(const float* __restrict__ in,
                                     f16* __restrict__ out, int K, int N) {
  int idx = blockIdx.x * blockDim.x + threadIdx.x;
  int total = K * N;
  if (idx >= total) return;
  int n = idx / K;
  int k = idx - n * K;
  out[idx] = (f16)in[(size_t)k * N + n];
}

// ---------------------------------------------------------------------------
// Kernel 2: LayerNorm, one block (256 thr) per row of 768, output f16
// ---------------------------------------------------------------------------
__global__ __launch_bounds__(256) void layernorm_kernel(
    const float* __restrict__ x, const float* __restrict__ g,
    const float* __restrict__ b, f16* __restrict__ xn) {
  const int row = blockIdx.x;
  const int tid = threadIdx.x;
  const float* xr = x + (size_t)row * DIM;

  float v0 = xr[tid], v1 = xr[tid + 256], v2 = xr[tid + 512];
  float s = v0 + v1 + v2;
#pragma unroll
  for (int m = 16; m >= 1; m >>= 1) s += __shfl_xor(s, m, 32);

  __shared__ float red[8];
  const int wv = tid >> 5, ln = tid & 31;
  if (ln == 0) red[wv] = s;
  __syncthreads();
  float tot = 0.0f;
#pragma unroll
  for (int i = 0; i < 8; ++i) tot += red[i];
  const float mean = tot * (1.0f / (float)DIM);

  float d0 = v0 - mean, d1 = v1 - mean, d2 = v2 - mean;
  float ss = d0 * d0 + d1 * d1 + d2 * d2;
#pragma unroll
  for (int m = 16; m >= 1; m >>= 1) ss += __shfl_xor(ss, m, 32);
  __syncthreads();
  if (ln == 0) red[wv] = ss;
  __syncthreads();
  float tot2 = 0.0f;
#pragma unroll
  for (int i = 0; i < 8; ++i) tot2 += red[i];
  const float rstd = rsqrtf(tot2 * (1.0f / (float)DIM) + 1e-5f);

  f16* o = xn + (size_t)row * DIM;
  o[tid]       = (f16)(d0 * rstd * g[tid]       + b[tid]);
  o[tid + 256] = (f16)(d1 * rstd * g[tid + 256] + b[tid + 256]);
  o[tid + 512] = (f16)(d2 * rstd * g[tid + 512] + b[tid + 512]);
}

// ---------------------------------------------------------------------------
// Kernel 3/5: tiled WMMA GEMM  C[M][N] = A[M][K](f16) * Bt[N][K](f16) + bias
// Block: 256 thr (8 waves), tile 128x128, K step 32.
// Wave (wm = w>>1 in 0..3, wn = w&1): 32x64 region = 2x4 subtiles of 16x16.
// ---------------------------------------------------------------------------
template <bool F16OUT>
__global__ __launch_bounds__(256) void gemm_wmma_kernel(
    const f16* __restrict__ A, const f16* __restrict__ Bt,
    const float* __restrict__ bias, void* __restrict__ Cv,
    int M, int N, int K) {
  constexpr int LDT = 40;  // 32 + 8 pad, keeps 16B-aligned chunks (80B stride)
  __shared__ alignas(16) f16 As[128 * LDT];
  __shared__ alignas(16) f16 Bs[128 * LDT];

  const int tid = threadIdx.x;
  const int m0 = blockIdx.y * 128;
  const int n0 = blockIdx.x * 128;
  const int w  = tid >> 5;
  const int wm = w >> 1;
  const int wn = w & 1;

  v8f acc[2][4];
#pragma unroll
  for (int i = 0; i < 2; ++i)
#pragma unroll
    for (int j = 0; j < 4; ++j) acc[i][j] = vzero8f();

  for (int k0 = 0; k0 < K; k0 += 32) {
    __syncthreads();
#pragma unroll
    for (int i = 0; i < 2; ++i) {
      int idx = tid + i * 256;        // 512 chunks of 8 f16
      int r = idx >> 2;               // 0..127
      int c = (idx & 3) * 8;          // 0,8,16,24
      *(v8h*)&As[r * LDT + c] = *(const v8h*)&A[(size_t)(m0 + r) * K + k0 + c];
      *(v8h*)&Bs[r * LDT + c] = *(const v8h*)&Bt[(size_t)(n0 + r) * K + k0 + c];
      if (k0 + 32 < K) {
        __builtin_prefetch(&A[(size_t)(m0 + r) * K + k0 + 32 + c], 0, 1);
        __builtin_prefetch(&Bt[(size_t)(n0 + r) * K + k0 + 32 + c], 0, 1);
      }
    }
    __syncthreads();

    v16h af[2], bf[4];
#pragma unroll
    for (int sm = 0; sm < 2; ++sm)
      af[sm] = ld_frag_a(&As[(wm * 32 + sm * 16) * LDT], LDT);
#pragma unroll
    for (int sn = 0; sn < 4; ++sn)
      bf[sn] = ld_frag_b(&Bs[(wn * 64 + sn * 16) * LDT], LDT);
#pragma unroll
    for (int sm = 0; sm < 2; ++sm)
#pragma unroll
      for (int sn = 0; sn < 4; ++sn)
        acc[sm][sn] = wmma32(af[sm], bf[sn], acc[sm][sn]);
  }

  const int lane = tid & 31;
  const int ln = lane & 15;
  const int kh = lane >> 4;
#pragma unroll
  for (int sm = 0; sm < 2; ++sm) {
#pragma unroll
    for (int sn = 0; sn < 4; ++sn) {
      const int row0 = m0 + wm * 32 + sm * 16 + 8 * kh;
      const int col  = n0 + wn * 64 + sn * 16 + ln;
      const float bv = bias[col];
#pragma unroll
      for (int r = 0; r < 8; ++r) {
        const float val = acc[sm][sn][r] + bv;
        if (F16OUT)
          ((f16*)Cv)[(size_t)(row0 + r) * N + col] = (f16)val;
        else
          ((float*)Cv)[(size_t)(row0 + r) * N + col] = val;
      }
    }
  }
}

// ---------------------------------------------------------------------------
// Kernel 4: flash attention, one block = 64 query rows of one (batch, head).
// 128 thr = 4 waves; each wave owns 16 query rows. Online softmax, WMMA for
// both S = Q*K^T and O += P*V.
// ---------------------------------------------------------------------------
__global__ __launch_bounds__(128) void attn_kernel(const f16* __restrict__ qkv,
                                                   f16* __restrict__ attn_out) {
  constexpr int LDT = 72;  // 64 + 8 pad (144B row stride, 16B aligned)
  __shared__ alignas(16) f16 Qs[64 * LDT];
  __shared__ alignas(16) f16 Ks[64 * LDT];
  __shared__ alignas(16) f16 Vt[64 * LDT];          // transposed: [d][j]
  __shared__ alignas(16) f16 Ps[4 * 16 * LDT];      // per-wave 16x64

  const int tid  = threadIdx.x;
  const int w    = tid >> 5;
  const int lane = tid & 31;
  const int ln   = lane & 15;
  const int kh   = lane >> 4;

  const int it = blockIdx.x;             // query tile 0..31
  const int bh = blockIdx.y;             // 0..47
  const int b  = bh / HEADS;
  const int h  = bh - b * HEADS;
  const size_t rowbase = (size_t)b * SEQ;
  const int qcol = h * DHEAD;
  const int kcol = DIM + h * DHEAD;
  const int vcol = 2 * DIM + h * DHEAD;
  const int q0 = it * 64;

  // Load + scale Q tile (64x64)
#pragma unroll
  for (int i = 0; i < 4; ++i) {
    int idx = tid + i * 128;             // 512 chunks
    int r = idx >> 3;                    // 0..63
    int c = (idx & 7) * 8;               // 0..56
    v8h q = *(const v8h*)&qkv[(rowbase + q0 + r) * QKV_N + qcol + c];
#pragma unroll
    for (int e = 0; e < 8; ++e) q[e] = q[e] * (f16)0.125f;  // 1/sqrt(64)
    *(v8h*)&Qs[r * LDT + c] = q;
  }
  __syncthreads();

  v16h qf[2];
  qf[0] = ld_frag_a(&Qs[(w * 16) * LDT + 0], LDT);
  qf[1] = ld_frag_a(&Qs[(w * 16) * LDT + 32], LDT);

  float m_i[8], l_i[8];
  v8f o[4];
#pragma unroll
  for (int r = 0; r < 8; ++r) { m_i[r] = -3.0e38f; l_i[r] = 0.0f; }
#pragma unroll
  for (int st = 0; st < 4; ++st) o[st] = vzero8f();

  for (int j0 = 0; j0 < SEQ; j0 += 64) {
    __syncthreads();  // all waves done reading K/V/P of previous iteration
#pragma unroll
    for (int i = 0; i < 4; ++i) {
      int idx = tid + i * 128;
      int r = idx >> 3;                  // key row 0..63
      int c = (idx & 7) * 8;             // d chunk
      *(v8h*)&Ks[r * LDT + c] =
          *(const v8h*)&qkv[(rowbase + j0 + r) * QKV_N + kcol + c];
      v8h vv = *(const v8h*)&qkv[(rowbase + j0 + r) * QKV_N + vcol + c];
#pragma unroll
      for (int e = 0; e < 8; ++e) Vt[(c + e) * LDT + r] = vv[e];
    }
    __syncthreads();

    // S = Q * K^T : 4 subtiles of 16x16, K-dim = 64 (2 WMMAs each)
    v8f s[4];
#pragma unroll
    for (int sj = 0; sj < 4; ++sj) {
      v16h b0 = ld_frag_b(&Ks[(sj * 16) * LDT + 0], LDT);
      v16h b1 = ld_frag_b(&Ks[(sj * 16) * LDT + 32], LDT);
      v8f c0 = vzero8f();
      c0 = wmma32(qf[0], b0, c0);
      c0 = wmma32(qf[1], b1, c0);
      s[sj] = c0;
    }

    // online softmax: per-row (r + 8*kh) stats across 16-lane column groups
    float bm[8];
#pragma unroll
    for (int r = 0; r < 8; ++r)
      bm[r] = fmaxf(fmaxf(s[0][r], s[1][r]), fmaxf(s[2][r], s[3][r]));
#pragma unroll
    for (int m = 8; m >= 1; m >>= 1)
#pragma unroll
      for (int r = 0; r < 8; ++r) bm[r] = fmaxf(bm[r], __shfl_xor(bm[r], m, 16));

    float mnew[8], scl[8], rs[8];
#pragma unroll
    for (int r = 0; r < 8; ++r) {
      mnew[r] = fmaxf(m_i[r], bm[r]);
      scl[r]  = __expf(m_i[r] - mnew[r]);
      rs[r]   = 0.0f;
    }
#pragma unroll
    for (int sj = 0; sj < 4; ++sj) {
#pragma unroll
      for (int r = 0; r < 8; ++r) {
        float e = __expf(s[sj][r] - mnew[r]);
        rs[r] += e;
        Ps[(w * 16 + r + 8 * kh) * LDT + sj * 16 + ln] = (f16)e;
      }
    }
#pragma unroll
    for (int m = 8; m >= 1; m >>= 1)
#pragma unroll
      for (int r = 0; r < 8; ++r) rs[r] += __shfl_xor(rs[r], m, 16);
#pragma unroll
    for (int r = 0; r < 8; ++r) {
      l_i[r] = l_i[r] * scl[r] + rs[r];
      m_i[r] = mnew[r];
    }
#pragma unroll
    for (int st = 0; st < 4; ++st)
#pragma unroll
      for (int r = 0; r < 8; ++r) o[st][r] *= scl[r];

    __syncthreads();  // P stores visible for fragment loads

    // O += P * V  (A = P 16x64, B = Vt [d][j])
    v16h pf[2];
    pf[0] = ld_frag_a(&Ps[(w * 16) * LDT + 0], LDT);
    pf[1] = ld_frag_a(&Ps[(w * 16) * LDT + 32], LDT);
#pragma unroll
    for (int st = 0; st < 4; ++st) {
      v16h vb0 = ld_frag_b(&Vt[(st * 16) * LDT + 0], LDT);
      v16h vb1 = ld_frag_b(&Vt[(st * 16) * LDT + 32], LDT);
      o[st] = wmma32(pf[0], vb0, o[st]);
      o[st] = wmma32(pf[1], vb1, o[st]);
    }
  }

  // normalize and store (f16 into attn workspace, [ROWS][768])
#pragma unroll
  for (int st = 0; st < 4; ++st) {
    const int col = h * DHEAD + st * 16 + ln;
#pragma unroll
    for (int r = 0; r < 8; ++r) {
      const size_t row = rowbase + q0 + w * 16 + r + 8 * kh;
      attn_out[row * DIM + col] = (f16)(o[st][r] * (1.0f / l_i[r]));
    }
  }
}

// ---------------------------------------------------------------------------
// Host-side orchestration
// ---------------------------------------------------------------------------
extern "C" void kernel_launch(void* const* d_in, const int* in_sizes, int n_in,
                              void* d_out, int out_size, void* d_ws, size_t ws_size,
                              hipStream_t stream) {
  const float* x     = (const float*)d_in[0];
  const float* ln_g  = (const float*)d_in[1];
  const float* ln_b  = (const float*)d_in[2];
  const float* W_qkv = (const float*)d_in[3];
  const float* b_qkv = (const float*)d_in[4];
  const float* W_out = (const float*)d_in[5];
  const float* b_out = (const float*)d_in[6];
  float* out = (float*)d_out;

  char* ws = (char*)d_ws;
  f16* xn     = (f16*)ws; ws += (size_t)ROWS * DIM * sizeof(f16);
  f16* qkv    = (f16*)ws; ws += (size_t)ROWS * QKV_N * sizeof(f16);
  f16* attn   = (f16*)ws; ws += (size_t)ROWS * DIM * sizeof(f16);
  f16* wqkv_t = (f16*)ws; ws += (size_t)QKV_N * DIM * sizeof(f16);
  f16* wout_t = (f16*)ws;

  {
    int tot = DIM * QKV_N;
    transpose_f16_kernel<<<(tot + 255) / 256, 256, 0, stream>>>(W_qkv, wqkv_t, DIM, QKV_N);
  }
  {
    int tot = DIM * DIM;
    transpose_f16_kernel<<<(tot + 255) / 256, 256, 0, stream>>>(W_out, wout_t, DIM, DIM);
  }

  layernorm_kernel<<<ROWS, 256, 0, stream>>>(x, ln_g, ln_b, xn);

  gemm_wmma_kernel<true><<<dim3(QKV_N / 128, ROWS / 128), 256, 0, stream>>>(
      xn, wqkv_t, b_qkv, (void*)qkv, ROWS, QKV_N, DIM);

  attn_kernel<<<dim3(SEQ / 64, BATCH * HEADS), 128, 0, stream>>>(qkv, attn);

  gemm_wmma_kernel<false><<<dim3(DIM / 128, ROWS / 128), 256, 0, stream>>>(
      attn, wout_t, b_out, (void*)out, ROWS, DIM, DIM);
}
